// SentimentClassifier_29712583753907
// MI455X (gfx1250) — compile-verified, matrix operand order checked by reading, not run on hardware
//
#include <hip/hip_runtime.h>
#include <hip/hip_bf16.h>
#include <stdint.h>

typedef __attribute__((ext_vector_type(16))) _Float16 v16h;
typedef __attribute__((ext_vector_type(8)))  _Float16 v8h;
typedef __attribute__((ext_vector_type(8)))  float    v8f;

#define BSZ   64
#define TLEN  512
#define HDIM  512
#define DDIM  300
#define DPAD  320
#define G4H   2048   // 4*H
#define NBLK  16     // persistent blocks in recurrence

// ---------- helpers ----------

// Load a 16-half WMMA fragment slice for this lane: 8 halfs at p, 8 at p+16.
__device__ __forceinline__ v16h ld_frag(const _Float16* p) {
  union { v16h v; v8h h[2]; } u;
  u.h[0] = *(const v8h*)(p);
  u.h[1] = *(const v8h*)(p + 16);
  return u.v;
}

__device__ __forceinline__ float sigmoidf_(float x) {
  return 1.f / (1.f + __expf(-x));
}

// Async 16-byte DMA global -> LDS (no VGPR roundtrip), tracked by ASYNCcnt.
__device__ __forceinline__ void async_copy_b128(void* lds_dst, const void* gsrc) {
  unsigned ldsoff = (unsigned)(uintptr_t)lds_dst;
  asm volatile("global_load_async_to_lds_b128 %0, %1, off"
               :: "v"(ldsoff), "v"(gsrc) : "memory");
}
__device__ __forceinline__ void async_wait0() {
  asm volatile("s_wait_asynccnt 0x0" ::: "memory");
}

// Grid-wide barrier over NBLK co-resident blocks (monotonic counter).
// Acquire fence executed by ALL threads: the block's 8 waves span both CUs of
// the WGP (separate L0s), so a single-thread invalidate is not enough.
__device__ __forceinline__ void gbar(unsigned* bar, unsigned target) {
  __threadfence();
  __syncthreads();
  if (threadIdx.x == 0) {
    atomicAdd(bar, 1u);
    while (*(volatile unsigned*)bar < target) __builtin_amdgcn_s_sleep(8);
  }
  __syncthreads();
  __threadfence();
}

// ---------- pack kernels ----------

__global__ void embed_pack(const int* __restrict__ X, const float* __restrict__ emb,
                           _Float16* __restrict__ dst, int BT, int D, int Kp) {
  int stride = gridDim.x * blockDim.x;
  for (int i = blockIdx.x * blockDim.x + threadIdx.x; i < BT * Kp; i += stride) {
    int bt = i / Kp, k = i - bt * Kp;
    float v = (k < D) ? emb[(size_t)X[bt] * D + k] : 0.f;
    dst[i] = (_Float16)v;
  }
}

__global__ void conv_pack(const float* __restrict__ src, _Float16* __restrict__ dst,
                          int rows, int K, int Kp) {
  int stride = gridDim.x * blockDim.x;
  for (int i = blockIdx.x * blockDim.x + threadIdx.x; i < rows * Kp; i += stride) {
    int r = i / Kp, k = i - r * Kp;
    dst[i] = (_Float16)((k < K) ? src[(size_t)r * K + k] : 0.f);
  }
}

// ---------- big GEMM: C[M,N] = A[M,K] * W[N,K]^T + bias1 + bias2 ----------
// Register-blocked: one wave -> 4 M-tiles x 2 N-tiles (64x32 output).
// B fragment reused across 4 WMMAs, A fragment across 2 -> 1.5 b128 loads/WMMA.
__global__ void wmma_gemm_bias(const _Float16* __restrict__ A, const _Float16* __restrict__ W,
                               const float* __restrict__ bias1, const float* __restrict__ bias2,
                               float* __restrict__ C, int M, int N, int K) {
  int gw   = (int)((blockIdx.x * blockDim.x + threadIdx.x) >> 5);
  int lane = threadIdx.x & 31;
  int nsup = N >> 5;                       // super-columns of 32
  int sm = gw / nsup, sn = gw - sm * nsup;
  int m0 = sm * 64, n0 = sn * 32;
  if (m0 >= M) return;
  int ln = lane & 15;
  int xh = (lane >> 4) << 3;               // 0 for lanes 0-15, 8 for lanes 16-31

  const _Float16* arow[4];
#pragma unroll
  for (int mt = 0; mt < 4; ++mt) arow[mt] = A + (size_t)(m0 + mt * 16 + ln) * K;
  const _Float16* wrow[2];
#pragma unroll
  for (int nt = 0; nt < 2; ++nt) wrow[nt] = W + (size_t)(n0 + nt * 16 + ln) * K;

  v8f acc[4][2];
#pragma unroll
  for (int nt = 0; nt < 2; ++nt) {
    float bs = bias1[n0 + nt * 16 + ln] + bias2[n0 + nt * 16 + ln];
#pragma unroll
    for (int mt = 0; mt < 4; ++mt)
#pragma unroll
      for (int r = 0; r < 8; ++r) acc[mt][nt][r] = bs;
  }

  for (int k0 = 0; k0 < K; k0 += 32) {
    v16h bf[2];
#pragma unroll
    for (int nt = 0; nt < 2; ++nt) bf[nt] = ld_frag(wrow[nt] + k0 + xh);
#pragma unroll
    for (int mt = 0; mt < 4; ++mt) {
      v16h af = ld_frag(arow[mt] + k0 + xh);
#pragma unroll
      for (int nt = 0; nt < 2; ++nt)
        acc[mt][nt] = __builtin_amdgcn_wmma_f32_16x16x32_f16(false, af, false, bf[nt],
                                                             (short)0, acc[mt][nt],
                                                             false, false);
    }
  }

#pragma unroll
  for (int mt = 0; mt < 4; ++mt)
#pragma unroll
    for (int nt = 0; nt < 2; ++nt) {
      float* cp = C + (size_t)(m0 + mt * 16 + xh) * N + (size_t)(n0 + nt * 16 + ln);
#pragma unroll
      for (int r = 0; r < 8; ++r) cp[(size_t)r * N] = acc[mt][nt][r];
    }
}

// ---------- persistent LSTM recurrence ----------
// NBLK blocks x 256 threads. Block w owns h columns [32w,32w+32) and the 128
// gate rows {j, j+512, j+1024, j+1536}. xg: [B*T, 4H] f32 (row bt = b*T+t).
// hbuf: [64,512] f16 shared state (zeroed before launch). hseq: [B,T,H] f16.
// The block's Whh slice (128x512 f16 = 128 KB) is staged into LDS ONCE via
// async DMA and stays resident for all 512 steps (CDNA5: 320 KB LDS/WGP).
__global__ void lstm_persistent(const float* __restrict__ xg, const _Float16* __restrict__ Whh,
                                _Float16* __restrict__ hbuf, _Float16* __restrict__ hseq,
                                unsigned* __restrict__ bar) {
  __shared__ _Float16 hs[BSZ][HDIM + 8];    // h state staged per step (65 KB)
  __shared__ _Float16 wsl[128][HDIM + 8];   // resident Whh slice (130 KB)
  __shared__ float    gl[BSZ][128];         // gates for this block's columns (32 KB)
  __shared__ float    cs[BSZ][32];          // persistent cell state (8 KB)

  const int tid  = threadIdx.x;
  const int blk  = blockIdx.x;
  const int lane = tid & 31;
  const int wv   = tid >> 5;               // 8 waves
  const int ln   = lane & 15;
  const int xh   = (lane >> 4) << 3;

  for (int i = tid; i < BSZ * 32; i += 256) cs[i >> 5][i & 31] = 0.f;

  const int chunk = wv >> 1;                                   // 0..3 = i,f,g,o
  const int gcol  = (chunk << 5) + ((wv & 1) << 4) + ln;       // 0..127 local
  const int nrow  = (chunk << 9) + (blk << 5) + ((wv & 1) << 4) + ln;  // Whh row

  // stage the block's Whh slice into LDS once: local row lr = c*32+w maps to
  // global Whh row c*512 + blk*32 + w.
  for (int i = tid; i < 128 * (HDIM / 8); i += 256) {
    int lr  = i >> 6;              // 64 16-byte chunks per row
    int col = (i & 63) << 3;
    int c   = lr >> 5, w = lr & 31;
    int grow = (c << 9) + (blk << 5) + w;
    async_copy_b128(&wsl[lr][col], Whh + (size_t)grow * HDIM + col);
  }
  async_wait0();
  __syncthreads();

  unsigned phase = 0;
  for (int t = 0; t < TLEN; ++t) {
    // stage h(t-1) into LDS with async DMA (memory -> LDS, no VGPR roundtrip)
    for (int i = tid; i < (BSZ * HDIM) / 8; i += 256) {
      int row = i >> 6;
      int col = (i & 63) << 3;
      async_copy_b128(&hs[row][col], hbuf + row * HDIM + col);
    }
    async_wait0();
    gbar(bar, (++phase) * NBLK);   // all blocks done reading old h

    // gates = xg[:,t,:] + h @ Whh^T   (4 M-tiles per wave, shared B fragment)
    v8f acc[4];
    const float* xgp = xg + (size_t)t * G4H + nrow;
#pragma unroll
    for (int mt = 0; mt < 4; ++mt)
#pragma unroll
      for (int r = 0; r < 8; ++r)
        acc[mt][r] = xgp[(size_t)(mt * 16 + r + xh) * ((size_t)TLEN * G4H)];

    for (int k0 = 0; k0 < HDIM; k0 += 32) {
      v16h b = ld_frag(&wsl[gcol][k0 + xh]);       // B fragment from resident LDS
#pragma unroll
      for (int mt = 0; mt < 4; ++mt) {
        v16h a = ld_frag(&hs[mt * 16 + ln][k0 + xh]);
        acc[mt] = __builtin_amdgcn_wmma_f32_16x16x32_f16(false, a, false, b, (short)0,
                                                         acc[mt], false, false);
      }
    }
#pragma unroll
    for (int mt = 0; mt < 4; ++mt)
#pragma unroll
      for (int r = 0; r < 8; ++r)
        gl[mt * 16 + r + xh][gcol] = acc[mt][r];
    __syncthreads();

    // cell update for this block's 32 h columns
    for (int i = tid; i < BSZ * 32; i += 256) {
      int b = i >> 5, jj = i & 31;
      float iv = sigmoidf_(gl[b][jj]);
      float fv = sigmoidf_(gl[b][32 + jj]);
      float gv = tanhf(gl[b][64 + jj]);
      float ov = sigmoidf_(gl[b][96 + jj]);
      float c  = fv * cs[b][jj] + iv * gv;
      cs[b][jj] = c;
      float h = ov * tanhf(c);
      int j = (blk << 5) + jj;
      hbuf[b * HDIM + j] = (_Float16)h;
      hseq[((size_t)b * TLEN + t) * HDIM + j] = (_Float16)h;
    }
    gbar(bar, (++phase) * NBLK);   // all blocks done writing new h
  }
}

// ---------- pooling + MLP head ----------
__global__ void pool_mlp(const _Float16* __restrict__ h2,
                         const float* __restrict__ W1, const float* __restrict__ b1,
                         const float* __restrict__ W2, const float* __restrict__ b2,
                         const float* __restrict__ W3, const float* __restrict__ b3,
                         float* __restrict__ out) {
  __shared__ float conc[2 * HDIM];
  __shared__ float z1[64];
  __shared__ float z2[16];
  int b = blockIdx.x, tid = threadIdx.x;

  for (int j = tid; j < HDIM; j += 256) {
    const _Float16* pp = h2 + ((size_t)b * TLEN) * HDIM + j;
    float s = 0.f, mx = -3.4e38f;
    for (int t = 0; t < TLEN; ++t) {
      float v = (float)pp[(size_t)t * HDIM];
      s += v;
      mx = fmaxf(mx, v);
    }
    conc[j]        = fmaxf(s * (1.f / TLEN), 0.f);  // relu(mean)
    conc[HDIM + j] = fmaxf(mx, 0.f);                // relu(max)
  }
  __syncthreads();
  if (tid < 64) {
    float a = b1[tid];
    const float* w = W1 + (size_t)tid * 2 * HDIM;
    for (int k = 0; k < 2 * HDIM; ++k) a += w[k] * conc[k];
    z1[tid] = fmaxf(a, 0.f);
  }
  __syncthreads();
  if (tid < 16) {
    float a = b2[tid];
    const float* w = W2 + (size_t)tid * 64;
    for (int k = 0; k < 64; ++k) a += w[k] * z1[k];
    z2[tid] = fmaxf(a, 0.f);
  }
  __syncthreads();
  if (tid == 0) {
    float a = b3[0];
    for (int k = 0; k < 16; ++k) a += W3[k] * z2[k];
    out[b] = 1.f / (1.f + __expf(-a));
  }
}

// ---------- launch ----------
extern "C" void kernel_launch(void* const* d_in, const int* in_sizes, int n_in,
                              void* d_out, int out_size, void* d_ws, size_t ws_size,
                              hipStream_t stream) {
  (void)in_sizes; (void)n_in; (void)out_size; (void)ws_size;

  const int*   X    = (const int*)  d_in[0];
  const float* emb  = (const float*)d_in[1];
  const float* Wih0 = (const float*)d_in[2];
  const float* Whh0 = (const float*)d_in[3];
  const float* bih0 = (const float*)d_in[4];
  const float* bhh0 = (const float*)d_in[5];
  const float* Wih1 = (const float*)d_in[6];
  const float* Whh1 = (const float*)d_in[7];
  const float* bih1 = (const float*)d_in[8];
  const float* bhh1 = (const float*)d_in[9];
  const float* W1   = (const float*)d_in[10];
  const float* b1   = (const float*)d_in[11];
  const float* W2   = (const float*)d_in[12];
  const float* b2   = (const float*)d_in[13];
  const float* W3   = (const float*)d_in[14];
  const float* b3   = (const float*)d_in[15];
  float* out = (float*)d_out;

  char* p = (char*)d_ws;
  auto take = [&](size_t bytes) {
    char* r = p;
    p += (bytes + 255) & ~(size_t)255;
    return r;
  };
  const size_t BT = (size_t)BSZ * TLEN;
  _Float16* Xf    = (_Float16*)take(BT * DPAD * 2);
  _Float16* Wih0h = (_Float16*)take((size_t)G4H * DPAD * 2);
  _Float16* Whh0h = (_Float16*)take((size_t)G4H * HDIM * 2);
  _Float16* Wih1h = (_Float16*)take((size_t)G4H * HDIM * 2);
  _Float16* Whh1h = (_Float16*)take((size_t)G4H * HDIM * 2);
  float*    xgb   = (float*)   take(BT * G4H * 4);
  _Float16* h1    = (_Float16*)take(BT * HDIM * 2);
  _Float16* h2    = (_Float16*)take(BT * HDIM * 2);
  _Float16* hbuf  = (_Float16*)take((size_t)BSZ * HDIM * 2);
  unsigned* bar   = (unsigned*)take(256);

  // pack inputs/weights to f16 (K padded to 32-multiples)
  embed_pack<<<4096, 256, 0, stream>>>(X, emb, Xf, (int)BT, DDIM, DPAD);
  conv_pack<<<1024, 256, 0, stream>>>(Wih0, Wih0h, G4H, DDIM, DPAD);
  conv_pack<<<1024, 256, 0, stream>>>(Whh0, Whh0h, G4H, HDIM, HDIM);
  conv_pack<<<1024, 256, 0, stream>>>(Wih1, Wih1h, G4H, HDIM, HDIM);
  conv_pack<<<1024, 256, 0, stream>>>(Whh1, Whh1h, G4H, HDIM, HDIM);

  // waves = (M/64) * (N/32); 8 waves per block
  const int gemm_blocks = ((int)(BT / 64) * (G4H / 32)) / 8;

  // layer 0
  wmma_gemm_bias<<<gemm_blocks, 256, 0, stream>>>(Xf, Wih0h, bih0, bhh0, xgb,
                                                  (int)BT, G4H, DPAD);
  hipMemsetAsync(hbuf, 0, (size_t)BSZ * HDIM * 2, stream);
  hipMemsetAsync(bar, 0, 256, stream);
  lstm_persistent<<<NBLK, 256, 0, stream>>>(xgb, Whh0h, hbuf, h1, bar);

  // layer 1
  wmma_gemm_bias<<<gemm_blocks, 256, 0, stream>>>(h1, Wih1h, bih1, bhh1, xgb,
                                                  (int)BT, G4H, HDIM);
  hipMemsetAsync(hbuf, 0, (size_t)BSZ * HDIM * 2, stream);
  hipMemsetAsync(bar, 0, 256, stream);
  lstm_persistent<<<NBLK, 256, 0, stream>>>(xgb, Whh1h, hbuf, h2, bar);

  // head
  pool_mlp<<<BSZ, 256, 0, stream>>>(h2, W1, b1, W2, b2, W3, b3, out);
}
